// TBPPDecodeAndCrop_59313498358308
// MI455X (gfx1250) — matrix operand order repaired and checked by reading, not compiled, not run on hardware
//
#include <hip/hip_runtime.h>
#include <hip/hip_bf16.h>
#include <math.h>

#define Bn   8
#define Hh   512
#define Ww   512
#define Nn   76800
#define TOPK 200
#define OUTH 32
#define OUTW 256
#define CONF_THR 0.01f
#define IOU_THR  0.45f
#define PAD_Di   16
#define PAD_Pf   1.6f

typedef __attribute__((ext_vector_type(16))) _Float16 v16h;
typedef __attribute__((ext_vector_type(8)))  float    v8f;

// ---------------------------------------------------------------------------
// K1: decode prior boxes + confidence threshold ("work" array for NMS)
// ---------------------------------------------------------------------------
__global__ void k_decode(const float* __restrict__ yp, const float* __restrict__ pxy,
                         const float* __restrict__ pwh, const float* __restrict__ pvar,
                         float* __restrict__ boxes, float* __restrict__ work) {
  int t = blockIdx.x * blockDim.x + threadIdx.x;
  if (t >= Bn * Nn) return;
  int n = t % Nn;
  const float* y = yp + (size_t)t * 19;
  float vx = pvar[n*4+0], vy = pvar[n*4+1], vw = pvar[n*4+2], vh = pvar[n*4+3];
  float px = pxy[n*2+0],  py = pxy[n*2+1];
  float pw = pwh[n*2+0],  ph = pwh[n*2+1];
  float bx = px + y[0] * vx * pw;
  float by = py + y[1] * vy * ph;
  float bw = pw * __expf(y[2] * vw);
  float bh = ph * __expf(y[3] * vh);
  boxes[(size_t)t*4+0] = bx - bw * 0.5f;
  boxes[(size_t)t*4+1] = by - bh * 0.5f;
  boxes[(size_t)t*4+2] = bx + bw * 0.5f;
  boxes[(size_t)t*4+3] = by + bh * 0.5f;
  float s = y[18];
  work[t] = (s > CONF_THR) ? s : -1.0f;
}

// ---------------------------------------------------------------------------
// K2: grayscale via WMMA. One wave = one 16x16 pixel tile = one
// v_wmma_f32_16x16x32_f16 producing 256 distinct gray values.
//   A[m][2n]   = r*wr + g*wg   (pixel m,n of the tile)
//   A[m][2n+1] = b
//   B[2n][n]=1, B[2n+1][n]=wb  ->  D[m][n] = gray(m,n)
// A layout (ISA 16-bit A 16x32): lane L (M=L%16):
//   L<16 : elems 0..7 <-> K=0..7,  elems 8..15 <-> K=16..23
//   L>=16: elems 0..7 <-> K=8..15, elems 8..15 <-> K=24..31
// B layout (ISA 16-bit B 32x16): lane L holds column N=L%16,
//   same K order as A: elem e <-> K = 8h+e (e<8), 8+8h+e (e>=8), h=L/16.
// ---------------------------------------------------------------------------
__global__ void k_gray_wmma(const float* __restrict__ img, float* __restrict__ gray) {
  const float WR = 0.2989f, WG = 0.587f, WB = 0.114f;
  int wave = (blockIdx.x * blockDim.x + threadIdx.x) >> 5;
  int lane = threadIdx.x & 31;
  const int tpi = (Hh/16) * (Ww/16);              // tiles per image
  int b  = wave / tpi;
  int r  = wave % tpi;
  int ty = (r / (Ww/16)) * 16;
  int tx = (r % (Ww/16)) * 16;
  int m  = lane & 15;
  int hi = lane >> 4;
  int n0 = hi * 4;                                // first pixel column of group 0

  const float* row = img + (((size_t)b * Hh + ty + m) * Ww + tx) * 3;
  float p0[12], p1[12];
  {
    const float4* g0 = (const float4*)(row + 3 * n0);          // 16B aligned
    ((float4*)p0)[0] = g0[0]; ((float4*)p0)[1] = g0[1]; ((float4*)p0)[2] = g0[2];
    const float4* g1 = (const float4*)(row + 3 * (n0 + 8));
    ((float4*)p1)[0] = g1[0]; ((float4*)p1)[1] = g1[1]; ((float4*)p1)[2] = g1[2];
  }
  v16h a;
  #pragma unroll
  for (int j = 0; j < 4; ++j) {
    a[2*j]     = (_Float16)(p0[3*j] * WR + p0[3*j+1] * WG);
    a[2*j+1]   = (_Float16)p0[3*j+2];
    a[8+2*j]   = (_Float16)(p1[3*j] * WR + p1[3*j+1] * WG);
    a[8+2*j+1] = (_Float16)p1[3*j+2];
  }
  v16h bm;
  int ncol = m, kbase = hi * 8;
  #pragma unroll
  for (int e = 0; e < 16; ++e) {
    int K = kbase + e + (e >= 8 ? 8 : 0);
    bm[e] = (K == 2*ncol) ? (_Float16)1.0f
          : (K == 2*ncol + 1) ? (_Float16)WB : (_Float16)0.0f;
  }
  v8f c = {};
  c = __builtin_amdgcn_wmma_f32_16x16x32_f16(false, a, false, bm, (short)0, c, false, false);

  // D layout: lane l -> column N=l%16; VGPR r -> row M = r + 8*(l>>4)
  float* gb = gray + (size_t)b * Hh * Ww;
  #pragma unroll
  for (int rr = 0; rr < 8; ++rr)
    gb[(ty + rr + 8*hi) * Ww + tx + ncol] = c[rr];
}

// ---------------------------------------------------------------------------
// K3: NMS, one block per batch. work/boxes stay L2-resident (12MB << 192MB).
// ---------------------------------------------------------------------------
__global__ void __launch_bounds__(512) k_nms(const float* __restrict__ boxes,
                                             float* __restrict__ work,
                                             int* __restrict__ idxs,
                                             int* __restrict__ valid) {
  int b = blockIdx.x;
  const float* bx = boxes + (size_t)b * Nn * 4;
  float* wk = work + (size_t)b * Nn;
  __shared__ float sv[512];
  __shared__ int   si[512];
  __shared__ float sb[4];
  __shared__ int   sdone;
  int t = threadIdx.x;

  for (int k = 0; k < TOPK; ++k) {
    float best = -2.0f; int bi = Nn;
    for (int i = t; i < Nn; i += 512) {
      float v = wk[i];
      if (v > best) { best = v; bi = i; }          // first-max within stride
    }
    sv[t] = best; si[t] = bi;
    __syncthreads();
    for (int s = 256; s > 0; s >>= 1) {
      if (t < s) {
        float ov = sv[t+s]; int oi = si[t+s];
        if (ov > sv[t] || (ov == sv[t] && oi < si[t])) { sv[t] = ov; si[t] = oi; }
      }
      __syncthreads();
    }
    if (t == 0) {
      float bv = sv[0];
      int ok = bv > 0.0f;
      int ii = ok ? si[0] : 0;                     // argmax of all -1 -> 0
      idxs[b*TOPK+k] = ii; valid[b*TOPK+k] = ok;
      sdone = !ok;
      if (ok) { sb[0]=bx[ii*4]; sb[1]=bx[ii*4+1]; sb[2]=bx[ii*4+2]; sb[3]=bx[ii*4+3]; }
    }
    __syncthreads();
    if (sdone) {
      if (t == 0)
        for (int kk = k + 1; kk < TOPK; ++kk) { idxs[b*TOPK+kk] = 0; valid[b*TOPK+kk] = 0; }
      break;
    }
    float a0=sb[0], a1=sb[1], a2=sb[2], a3=sb[3];
    float areaA = (a2 - a0) * (a3 - a1);
    for (int i = t; i < Nn; i += 512) {
      float c0 = bx[i*4], c1 = bx[i*4+1], c2 = bx[i*4+2], c3 = bx[i*4+3];
      float x1 = fmaxf(a0, c0), y1 = fmaxf(a1, c1);
      float x2 = fminf(a2, c2), y2 = fminf(a3, c3);
      float inter = fmaxf(x2 - x1, 0.f) * fmaxf(y2 - y1, 0.f);
      float areaB = (c2 - c0) * (c3 - c1);
      float iou = inter / (areaA + areaB - inter + 1e-9f);
      if (iou > IOU_THR) wk[i] = -1.f;             // self-IoU=1 suppresses chosen
    }
    __syncthreads();
  }
}

// ---------------------------------------------------------------------------
// K4: gather, "good" rows (elems 0..18), homography M via 8x8 Gauss-Jordan
// ---------------------------------------------------------------------------
__global__ void k_good(const float* __restrict__ yp, const float* __restrict__ pxy,
                       const float* __restrict__ pwh, const float* __restrict__ pvar,
                       const float* __restrict__ boxes, const int* __restrict__ idxs,
                       const int* __restrict__ valid, float* __restrict__ outBoxes,
                       float* __restrict__ Mh, int* __restrict__ cArr) {
  int t = blockIdx.x * blockDim.x + threadIdx.x;
  if (t >= Bn * TOPK) return;
  int b = t / TOPK;
  int idx = idxs[t];
  int vld = valid[t];
  float vf = vld ? 1.f : 0.f;

  const float* y = yp + ((size_t)b * Nn + idx) * 19;
  float vx = pvar[idx*4+0], vyv = pvar[idx*4+1], vw = pvar[idx*4+2], vh = pvar[idx*4+3];
  float px = pxy[idx*2+0],  py  = pxy[idx*2+1];
  float pw = pwh[idx*2+0],  ph  = pwh[idx*2+1];
  const float* gb = boxes + ((size_t)b * Nn + idx) * 4;
  (void)vw;

  float ref8[8] = { px-pw*0.5f, py-ph*0.5f, px+pw*0.5f, py-ph*0.5f,
                    px+pw*0.5f, py+ph*0.5f, px-pw*0.5f, py+ph*0.5f };
  float sx = pw * vx, sy = ph * vyv;
  float quads[8];
  #pragma unroll
  for (int j = 0; j < 8; ++j) quads[j] = ref8[j] + y[4+j] * ((j & 1) ? sy : sx);
  float rboxs[5] = { px + y[12]*sx, py + y[13]*sy, px + y[14]*sx, py + y[15]*sy,
                     __expf(y[16] * vh) * ph };

  float* g = outBoxes + (size_t)t * 20;
  g[0]=gb[0]*vf; g[1]=gb[1]*vf; g[2]=gb[2]*vf; g[3]=gb[3]*vf;
  #pragma unroll
  for (int j = 0; j < 8; ++j) g[4+j] = quads[j] * vf;
  #pragma unroll
  for (int j = 0; j < 5; ++j) g[12+j] = rboxs[j] * vf;
  g[17] = y[18] * vf;
  g[18] = vf;                                      // ones * vf ; g[19] in K5

  const float defp[8] = {8.f,8.f,72.f,8.f,72.f,40.f,8.f,40.f};
  float polys[8];
  #pragma unroll
  for (int j = 0; j < 8; ++j) polys[j] = vld ? quads[j] * 512.0f : defp[j];
  float u0=polys[0],v0=polys[1],u1=polys[2],v1=polys[3];
  float u2=polys[4],v2=polys[5],u3=polys[6],v3=polys[7];
  float boxh = (sqrtf((u0-u3)*(u0-u3)+(v0-v3)*(v0-v3)) +
                sqrtf((u1-u2)*(u1-u2)+(v1-v2)*(v1-v2))) * 0.5f;
  float boxw = (sqrtf((u0-u1)*(u0-u1)+(v0-v1)*(v0-v1)) +
                sqrtf((u3-u2)*(u3-u2)+(v3-v2)*(v3-v2))) * 0.5f;
  float wmod = fminf(fmaxf((float)OUTH * boxw / (boxh + 1e-6f), 0.f), (float)OUTW);
  cArr[t] = (int)wmod;

  float X[4] = {PAD_Pf, wmod - PAD_Pf, wmod - PAD_Pf, PAD_Pf};
  float Y[4] = {PAD_Pf, PAD_Pf, (float)OUTH - PAD_Pf, (float)OUTH - PAD_Pf};
  float U[4] = {u0,u1,u2,u3}, V[4] = {v0,v1,v2,v3};
  float A[8][9];
  #pragma unroll
  for (int j = 0; j < 4; ++j) {
    float* r0 = A[j];
    r0[0]=X[j]; r0[1]=Y[j]; r0[2]=1.f; r0[3]=0.f; r0[4]=0.f; r0[5]=0.f;
    r0[6]=-X[j]*U[j]; r0[7]=-Y[j]*U[j]; r0[8]=U[j];
    float* r1 = A[4+j];
    r1[0]=0.f; r1[1]=0.f; r1[2]=0.f; r1[3]=X[j]; r1[4]=Y[j]; r1[5]=1.f;
    r1[6]=-X[j]*V[j]; r1[7]=-Y[j]*V[j]; r1[8]=V[j];
  }
  for (int col = 0; col < 8; ++col) {              // Gauss-Jordan, partial pivot
    int piv = col; float mx = fabsf(A[col][col]);
    for (int rr = col + 1; rr < 8; ++rr) { float av = fabsf(A[rr][col]); if (av > mx) { mx = av; piv = rr; } }
    if (piv != col)
      for (int cc = col; cc < 9; ++cc) { float tp = A[col][cc]; A[col][cc] = A[piv][cc]; A[piv][cc] = tp; }
    float inv = 1.0f / A[col][col];
    for (int cc = col; cc < 9; ++cc) A[col][cc] *= inv;
    for (int rr = 0; rr < 8; ++rr) {
      if (rr == col) continue;
      float f = A[rr][col];
      for (int cc = col; cc < 9; ++cc) A[rr][cc] -= f * A[col][cc];
    }
  }
  #pragma unroll
  for (int j = 0; j < 8; ++j) Mh[(size_t)t * 8 + j] = A[j][8];
}

// ---------------------------------------------------------------------------
// K5: per-batch serial cumsum -> starts + offsets (tiny: 8 threads)
// ---------------------------------------------------------------------------
__global__ void k_cumsum(const int* __restrict__ cArr, const int* __restrict__ valid,
                         int* __restrict__ starts, float* __restrict__ outBoxes) {
  int b = threadIdx.x;
  if (b >= Bn) return;
  float cum = 0.f;
  for (int k = 0; k < TOPK; ++k) {
    int t = b * TOPK + k;
    int vld = valid[t];
    float inc = vld ? (float)(cArr[t] + PAD_Di) : 0.f;
    starts[t] = (int)cum;                          // == cum_after - inc
    cum += inc;
    float off = vld ? (cum - (float)PAD_Di * 0.5f) : 0.f;
    outBoxes[(size_t)t * 20 + 19] = off * (vld ? 1.f : 0.f);
  }
}

// ---------------------------------------------------------------------------
// K6: owner map: intervals [start_k, start_k+c_k) are disjoint -> <=1 owner/col
// ---------------------------------------------------------------------------
__global__ void k_owner(const int* __restrict__ cArr, const int* __restrict__ valid,
                        const int* __restrict__ starts, int* __restrict__ owner,
                        int* __restrict__ relA) {
  int t = blockIdx.x * blockDim.x + threadIdx.x;
  if (t >= Bn * OUTW) return;
  int b = t / OUTW, j = t % OUTW;
  int o = -1, rl = 0;
  for (int k = 0; k < TOPK; ++k) {
    int q = b * TOPK + k;
    if (!valid[q]) continue;
    int r = j - starts[q];
    if (r >= 0 && r < cArr[q]) { o = k; rl = r; break; }
  }
  owner[t] = o; relA[t] = rl;
}

// ---------------------------------------------------------------------------
// K7: fused perspective warp + bilinear sample + strip assembly -> crops
// ---------------------------------------------------------------------------
__global__ void k_crop(const float* __restrict__ gray, const float* __restrict__ Mh,
                       const int* __restrict__ owner, const int* __restrict__ relA,
                       float* __restrict__ crops) {
  int t = blockIdx.x * blockDim.x + threadIdx.x;
  if (t >= Bn * OUTW * OUTH) return;
  int i = t % OUTH;
  int j = (t / OUTH) % OUTW;
  int b = t / (OUTH * OUTW);
  int k = owner[b * OUTW + j];
  float out = 0.f;
  if (k >= 0) {
    const float* m = Mh + ((size_t)(b * TOPK + k)) * 8;
    float xo = (float)relA[b * OUTW + j];
    float yo = (float)i;
    float den = m[6] * xo + m[7] * yo + 1.0f;
    float u = (m[0] * xo + m[1] * yo + m[2]) / den;
    float v = (m[3] * xo + m[4] * yo + m[5]) / den;
    const float* g = gray + (size_t)b * Hh * Ww;
    float u0f = floorf(u), v0f = floorf(v);
    float du = u - u0f, dv = v - v0f;
    int u0 = (int)u0f, v0 = (int)v0f;
    auto tap = [&](int vi, int ui, float w) -> float {
      bool inb = (vi >= 0) && (vi < Hh) && (ui >= 0) && (ui < Ww);
      int vc = vi < 0 ? 0 : (vi > Hh - 1 ? Hh - 1 : vi);
      int uc = ui < 0 ? 0 : (ui > Ww - 1 ? Ww - 1 : ui);
      float val = g[vc * Ww + uc];
      return inb ? val * w : 0.f;
    };
    out = tap(v0,   u0,   (1.f-dv)*(1.f-du)) + tap(v0,   u0+1, (1.f-dv)*du)
        + tap(v0+1, u0,   dv*(1.f-du))       + tap(v0+1, u0+1, dv*du);
  }
  crops[t] = out;                                  // (b, j, i) -> b*8192 + j*32 + i
}

// ---------------------------------------------------------------------------
extern "C" void kernel_launch(void* const* d_in, const int* in_sizes, int n_in,
                              void* d_out, int out_size, void* d_ws, size_t ws_size,
                              hipStream_t stream) {
  const float* images = (const float*)d_in[0];
  const float* y_pred = (const float*)d_in[1];
  const float* pxy    = (const float*)d_in[2];
  const float* pwh    = (const float*)d_in[3];
  const float* pvar   = (const float*)d_in[4];

  float* crops    = (float*)d_out;                          // (8,256,32,1)
  float* outBoxes = crops + (size_t)Bn * OUTW * OUTH;       // (8,200,20)

  char* w = (char*)d_ws;
  auto alloc = [&](size_t bytes) { void* p = (void*)w; w += (bytes + 255) & ~(size_t)255; return p; };
  float* boxes  = (float*)alloc((size_t)Bn * Nn * 4 * sizeof(float));
  float* work   = (float*)alloc((size_t)Bn * Nn * sizeof(float));
  float* gray   = (float*)alloc((size_t)Bn * Hh * Ww * sizeof(float));
  int*   idxs   = (int*)alloc((size_t)Bn * TOPK * sizeof(int));
  int*   valid  = (int*)alloc((size_t)Bn * TOPK * sizeof(int));
  float* Mh     = (float*)alloc((size_t)Bn * TOPK * 8 * sizeof(float));
  int*   cArr   = (int*)alloc((size_t)Bn * TOPK * sizeof(int));
  int*   starts = (int*)alloc((size_t)Bn * TOPK * sizeof(int));
  int*   owner  = (int*)alloc((size_t)Bn * OUTW * sizeof(int));
  int*   relA   = (int*)alloc((size_t)Bn * OUTW * sizeof(int));

  k_decode<<<(Bn * Nn + 255) / 256, 256, 0, stream>>>(y_pred, pxy, pwh, pvar, boxes, work);
  k_gray_wmma<<<(Bn * (Hh/16) * (Ww/16) * 32) / 256, 256, 0, stream>>>(images, gray);
  k_nms<<<Bn, 512, 0, stream>>>(boxes, work, idxs, valid);
  k_good<<<(Bn * TOPK + 255) / 256, 256, 0, stream>>>(y_pred, pxy, pwh, pvar, boxes,
                                                      idxs, valid, outBoxes, Mh, cArr);
  k_cumsum<<<1, Bn, 0, stream>>>(cArr, valid, starts, outBoxes);
  k_owner<<<(Bn * OUTW + 255) / 256, 256, 0, stream>>>(cArr, valid, starts, owner, relA);
  k_crop<<<(Bn * OUTW * OUTH + 255) / 256, 256, 0, stream>>>(gray, Mh, owner, relA, crops);
}
